// VolumetricTriangulationNet_24498493456406
// MI455X (gfx1250) — compile-verified, matrix operand order checked by reading, not run on hardware
//
#include <hip/hip_runtime.h>
#include <string>
#include <unordered_map>
#include <cstddef>

// ---------------------------------------------------------------------------
// Sizes from the reference
#define NC      6
#define CINF    256
#define FHW     128
#define CMID    32
#define NVOXS   32
#define NVTOT   (NVOXS*NVOXS*NVOXS)   // 32768
#define NJ      20
#define BN_EPS  1e-5f

typedef __attribute__((ext_vector_type(16))) _Float16 v16h;
typedef __attribute__((ext_vector_type(8)))  float    v8f;

__device__ __forceinline__ int iclamp(int v, int lo, int hi) {
  return v < lo ? lo : (v > hi ? hi : v);
}

// ---------------------------------------------------------------------------
// Generic conv3d (+fused BN scale/bias, optional residual add, optional ReLU)
// as an implicit-im2col GEMM on v_wmma_f32_16x16x32_f16.
//   in  : [Cin, D,H,W]  f32
//   wgt : [Cout, Cin*KSZ^3] f32 (natural OIDHW flattening)
//   sb  : scale[0..Cout), bias[Cout..2*Cout)
//   out : [Cout, D,H,W] f32
//
// Requires Kdim = Cin*KSZ^3 to be a multiple of 8 (true for every layer here),
// so each 8-wide K-group of a 32-chunk is either fully in- or out-of-bounds.
// All gathers are branch-free: addresses are clamped in-range, values zeroed
// by selects, so loads issue back-to-back with a single wait.
//
// Block = 128 threads = 4 waves; each wave computes a 16(Cout) x 32(voxel)
// tile = two WMMAs per K-chunk sharing one A fragment.
//
// CDNA5 16-bit A-matrix per-lane K mapping (ISA 7.12.2):
//   lanes 0-15 : elements 0..7 -> K=kk+0..7,   8..15 -> K=kk+16..23
//   lanes16-31 : elements 0..7 -> K=kk+8..15,  8..15 -> K=kk+24..31
// B assumed symmetric with N in place of M.
template<int KSZ>
__global__ __launch_bounds__(128)
void conv3d_bn_act_wmma(const float* __restrict__ in,
                        const float* __restrict__ wgt,
                        const float* __restrict__ sb,
                        const float* __restrict__ addsrc,
                        float* __restrict__ out,
                        int Cin, int Cout, int D, int H, int W,
                        int pad, int relu)
{
  const int K3   = KSZ * KSZ * KSZ;
  const int Kdim = Cin * K3;
  const int N    = D * H * W;

  const int lane   = threadIdx.x & 31;
  const int wave   = threadIdx.x >> 5;
  const int laneM  = lane & 15;
  const int laneHi = lane >> 4;

  const int n0 = blockIdx.x * 128 + wave * 32;  // voxel tile base (2 subtiles)
  const int m0 = blockIdx.y * 16;               // cout tile base

  // Column voxels for the two N-subtiles handled by this wave.
  int  colv[2];
  bool cval[2];
  int  vzv[2], vyv[2], vxv[2];
#pragma unroll
  for (int s = 0; s < 2; ++s) {
    const int col = n0 + s * 16 + laneM;
    const bool cv = col < N;
    const int nc  = cv ? col : 0;
    colv[s] = col; cval[s] = cv;
    vxv[s] = nc % W;
    const int t = nc / W;
    vyv[s] = t % H;
    vzv[s] = t / H;
  }

  const int  mrow   = m0 + laneM;
  const bool mvalid = mrow < Cout;
  const float* wrow = wgt + (size_t)(mvalid ? mrow : 0) * Kdim;

  v8f acc0 = {0.f, 0.f, 0.f, 0.f, 0.f, 0.f, 0.f, 0.f};
  v8f acc1 = {0.f, 0.f, 0.f, 0.f, 0.f, 0.f, 0.f, 0.f};

  for (int kk = 0; kk < Kdim; kk += 32) {
    const int  b0 = kk + (laneHi << 3);        // K-group for elements 0..7
    const int  b1 = b0 + 16;                   // K-group for elements 8..15
    const bool g0 = (b0 + 8) <= Kdim;          // whole-group validity
    const bool g1 = (b1 + 8) <= Kdim;

    // ---- A: 16x32 weight tile; rows contiguous -> 4 aligned float4 loads.
    const int c0 = g0 ? b0 : 0;
    const int c1 = g1 ? b1 : 0;
    const float4 wa = *(const float4*)(wrow + c0);
    const float4 wb = *(const float4*)(wrow + c0 + 4);
    const float4 wc = *(const float4*)(wrow + c1);
    const float4 wd = *(const float4*)(wrow + c1 + 4);
    const bool a0 = mvalid && g0;
    const bool a1 = mvalid && g1;
    float av[16] = {wa.x, wa.y, wa.z, wa.w, wb.x, wb.y, wb.z, wb.w,
                    wc.x, wc.y, wc.z, wc.w, wd.x, wd.y, wd.z, wd.w};
    v16h a;
#pragma unroll
    for (int e = 0; e < 16; ++e) {
      const bool ok = (e < 8) ? a0 : a1;
      a[e] = (_Float16)(ok ? av[e] : 0.f);
    }

    // ---- B: two 32x16 im2col tiles, branch-free clamped gathers.
    v16h bfr[2];
#pragma unroll
    for (int s = 0; s < 2; ++s) {
      const int vz = vzv[s], vy = vyv[s], vx = vxv[s];
      const bool cv = cval[s];
      float vals[16];
#pragma unroll
      for (int e = 0; e < 16; ++e) {
        const int  kidx = (e < 8) ? (b0 + e) : (b1 + (e - 8));
        const bool gv   = ((e < 8) ? g0 : g1) && cv;
        const int  kc   = gv ? kidx : 0;
        const int  c    = kc / K3;
        const int  r    = kc % K3;
        const int  dz   = r / (KSZ * KSZ);
        const int  rr   = r % (KSZ * KSZ);
        const int  dy   = rr / KSZ;
        const int  dx   = rr % KSZ;
        const int  z = vz + dz - pad;
        const int  y = vy + dy - pad;
        const int  x = vx + dx - pad;
        const bool inb = (z >= 0) & (z < D) & (y >= 0) & (y < H) &
                         (x >= 0) & (x < W);
        const int  zc = iclamp(z, 0, D - 1);
        const int  yc = iclamp(y, 0, H - 1);
        const int  xc = iclamp(x, 0, W - 1);
        const float v = in[((size_t)c * D + zc) * H * W + (size_t)yc * W + xc];
        vals[e] = (gv && inb) ? v : 0.f;
      }
      v16h b;
#pragma unroll
      for (int e = 0; e < 16; ++e) b[e] = (_Float16)vals[e];
      bfr[s] = b;
    }

    acc0 = __builtin_amdgcn_wmma_f32_16x16x32_f16(false, a, false, bfr[0],
                                                  (short)0, acc0, false, false);
    acc1 = __builtin_amdgcn_wmma_f32_16x16x32_f16(false, a, false, bfr[1],
                                                  (short)0, acc1, false, false);
  }

  // ---- Epilogue: fused BN scale/bias, optional residual add, optional ReLU.
#pragma unroll
  for (int s = 0; s < 2; ++s) {
    const v8f acc = s ? acc1 : acc0;
    const int col = colv[s];
    if (cval[s]) {
#pragma unroll
      for (int r = 0; r < 8; ++r) {
        const int m = m0 + r + laneHi * 8;
        if (m < Cout) {
          float v = acc[r] * sb[m] + sb[Cout + m];
          if (addsrc) v += addsrc[(size_t)m * N + col];
          if (relu)   v = v > 0.f ? v : 0.f;
          out[(size_t)m * N + col] = v;
        }
      }
    }
  }
}

// ---------------------------------------------------------------------------
// Fold conv bias + BN into per-channel scale/bias:  y = conv*s + b'
__global__ void fuse_bn_kernel(const float* __restrict__ cb,
                               const float* __restrict__ g,
                               const float* __restrict__ bt,
                               const float* __restrict__ m,
                               const float* __restrict__ v,
                               float* __restrict__ sb, int C, int has_bn)
{
  int i = blockIdx.x * blockDim.x + threadIdx.x;
  if (i >= C) return;
  if (has_bn) {
    float s = g[i] * rsqrtf(v[i] + BN_EPS);
    sb[i]     = s;
    sb[C + i] = (cb[i] - m[i]) * s + bt[i];
  } else {
    sb[i]     = 1.f;
    sb[C + i] = cb[i];
  }
}

// ---------------------------------------------------------------------------
// 2x2x2 max pool, stride 2
__global__ void maxpool2_kernel(const float* __restrict__ in,
                                float* __restrict__ out, int C, int S)
{
  const int So = S >> 1;
  const int No = So * So * So;
  const int total = C * No;
  int idx = blockIdx.x * blockDim.x + threadIdx.x;
  if (idx >= total) return;
  const int c = idx / No;
  const int r = idx % No;
  const int x = r % So;
  const int y = (r / So) % So;
  const int z = r / (So * So);
  float mx = -3.402823466e38f;
#pragma unroll
  for (int dz = 0; dz < 2; ++dz)
#pragma unroll
    for (int dy = 0; dy < 2; ++dy)
#pragma unroll
      for (int dx = 0; dx < 2; ++dx) {
        float v = in[(((size_t)c * S + (2*z+dz)) * S + (2*y+dy)) * S + (2*x+dx)];
        mx = fmaxf(mx, v);
      }
  out[idx] = mx;
}

// ---------------------------------------------------------------------------
// ConvTranspose3d k=2 s=2 (+bias, fused BN, ReLU, then add skip)
//   in [Cin,S^3], w [Cin,Cout,2,2,2], out/skip [Cout,(2S)^3]
__global__ void upconv_kernel(const float* __restrict__ in,
                              const float* __restrict__ w,
                              const float* __restrict__ sb,
                              const float* __restrict__ skip,
                              float* __restrict__ out,
                              int Cin, int Cout, int S)
{
  const int So = 2 * S;
  const int No = So * So * So;
  const int total = Cout * No;
  int idx = blockIdx.x * blockDim.x + threadIdx.x;
  if (idx >= total) return;
  const int co = idx / No;
  const int r  = idx % No;
  const int ox = r % So;
  const int oy = (r / So) % So;
  const int oz = r / (So * So);
  const int x = ox >> 1, k = ox & 1;
  const int y = oy >> 1, j = oy & 1;
  const int z = oz >> 1, i = oz & 1;
  const int wk = i * 4 + j * 2 + k;
  const int S3 = S * S * S;
  const int sp = (z * S + y) * S + x;
  float acc = 0.f;
  for (int ci = 0; ci < Cin; ++ci)
    acc += in[(size_t)ci * S3 + sp] * w[((size_t)ci * Cout + co) * 8 + wk];
  float v = acc * sb[co] + sb[Cout + co];
  v = v > 0.f ? v : 0.f;
  v += skip[idx];
  out[idx] = v;
}

// ---------------------------------------------------------------------------
// Unprojection with radial/tangential distortion + bilinear sampling.
// feat [NC,CMID,FHW,FHW], M [NC,4,3], K [NC,3,3], rd/td [NC,2], grid [NV,3]
// vols [NC,CMID,NV]
__global__ void unproject_kernel(const float* __restrict__ feat,
                                 const float* __restrict__ Mm,
                                 const float* __restrict__ Kc,
                                 const float* __restrict__ rd,
                                 const float* __restrict__ td,
                                 const float* __restrict__ grid,
                                 float* __restrict__ vols, int NV)
{
  int idx = blockIdx.x * blockDim.x + threadIdx.x;
  if (idx >= NC * NV) return;
  const int cam = idx / NV;
  const int vid = idx % NV;
  const float* M = Mm + cam * 12;
  const float gx = grid[vid * 3 + 0];
  const float gy = grid[vid * 3 + 1];
  const float gz = grid[vid * 3 + 2];
  const float p0 = gx * M[0] + gy * M[3] + gz * M[6] + M[9];
  const float p1 = gx * M[1] + gy * M[4] + gz * M[7] + M[10];
  const float p2 = gx * M[2] + gy * M[5] + gz * M[8] + M[11];
  float u = p0 / p2;
  float vv = p1 / p2;
  const float* K = Kc + cam * 9;
  const float fx = K[0], skew = K[3], fy = K[4], cx = K[6], cy = K[7];
  const float yN = (vv - cy) / fy;
  const float xN = (u - cx - skew * yN) / fx;
  const float r2 = xN * xN + yN * yN;
  const float alpha = rd[cam * 2] * r2 + rd[cam * 2 + 1] * r2 * r2;
  const float xy = xN * yN;
  const float t0 = td[cam * 2], t1 = td[cam * 2 + 1];
  const float dx = 2.f * t0 * xy + t1 * (r2 + 2.f * xN * xN);
  const float dy = t0 * (r2 + 2.f * yN * yN) + 2.f * t1 * xy;
  const float xd = xN * (1.f + alpha) + dx;
  const float yd = yN * (1.f + alpha) + dy;
  u  = xd * fx + cx + skew * yd;
  vv = yd * fy + cy;
  const float x0 = floorf(u), y0 = floorf(vv);
  const float wx = u - x0,   wy = vv - y0;
  const int x0i = (int)fminf(fmaxf(x0, 0.f), (float)(FHW - 1));
  const int x1i = (int)fminf(fmaxf(x0 + 1.f, 0.f), (float)(FHW - 1));
  const int y0i = (int)fminf(fmaxf(y0, 0.f), (float)(FHW - 1));
  const int y1i = (int)fminf(fmaxf(y0 + 1.f, 0.f), (float)(FHW - 1));
  const float w00 = (1.f - wx) * (1.f - wy);
  const float w01 = wx * (1.f - wy);
  const float w10 = (1.f - wx) * wy;
  const float w11 = wx * wy;
  const int o00 = y0i * FHW + x0i, o01 = y0i * FHW + x1i;
  const int o10 = y1i * FHW + x0i, o11 = y1i * FHW + x1i;
  const float* fb = feat + (size_t)cam * CMID * FHW * FHW;
  for (int c = 0; c < CMID; ++c) {
    const float* fc = fb + (size_t)c * FHW * FHW;
    vols[((size_t)cam * CMID + c) * NV + vid] =
        fc[o00] * w00 + fc[o01] * w01 + fc[o10] * w10 + fc[o11] * w11;
  }
}

// ---------------------------------------------------------------------------
// Softmax-weighted view fusion: per (c,v): w_n = softmax_n(vols); sum vols*w.
__global__ void fuse_views_kernel(const float* __restrict__ vols,
                                  float* __restrict__ vol, int NV)
{
  int idx = blockIdx.x * blockDim.x + threadIdx.x;
  if (idx >= CMID * NV) return;
  const int c = idx / NV;
  const int v = idx % NV;
  float mx = -3.402823466e38f;
  for (int n = 0; n < NC; ++n)
    mx = fmaxf(mx, vols[((size_t)n * CMID + c) * NV + v]);
  float se = 0.f, acc = 0.f;
  for (int n = 0; n < NC; ++n) {
    const float val = vols[((size_t)n * CMID + c) * NV + v];
    const float e = expf(val - mx);
    se  += e;
    acc += val * e;
  }
  vol[idx] = acc / se;
}

// ---------------------------------------------------------------------------
// coords[j] = sum_v softmax(hm[j])_v * grid[v]
__global__ __launch_bounds__(256)
void softargmax_kernel(const float* __restrict__ hm,
                       const float* __restrict__ grid,
                       float* __restrict__ coords, int NV)
{
  __shared__ float red[256];
  const int j = blockIdx.x;
  const int tid = threadIdx.x;
  const float* h = hm + (size_t)j * NV;
  float mx = -3.402823466e38f;
  for (int v = tid; v < NV; v += 256) mx = fmaxf(mx, h[v]);
  red[tid] = mx; __syncthreads();
  for (int s = 128; s > 0; s >>= 1) {
    if (tid < s) red[tid] = fmaxf(red[tid], red[tid + s]);
    __syncthreads();
  }
  mx = red[0]; __syncthreads();
  float se = 0.f, sx = 0.f, sy = 0.f, sz = 0.f;
  for (int v = tid; v < NV; v += 256) {
    const float e = expf(h[v] - mx);
    se += e;
    sx += e * grid[3 * v + 0];
    sy += e * grid[3 * v + 1];
    sz += e * grid[3 * v + 2];
  }
  float sums[4] = {se, sx, sy, sz};
  float tot[4];
  for (int q = 0; q < 4; ++q) {
    red[tid] = sums[q]; __syncthreads();
    for (int s = 128; s > 0; s >>= 1) {
      if (tid < s) red[tid] += red[tid + s];
      __syncthreads();
    }
    tot[q] = red[0]; __syncthreads();
  }
  if (tid == 0) {
    coords[j * 3 + 0] = tot[1] / tot[0];
    coords[j * 3 + 1] = tot[2] / tot[0];
    coords[j * 3 + 2] = tot[3] / tot[0];
  }
}

// ---------------------------------------------------------------------------
static void launch_conv(const float* in, const float* w, const float* sb,
                        const float* add, float* out,
                        int Cin, int Cout, int D, int H, int Wd,
                        int ksz, int pad, int relu, hipStream_t st)
{
  const int N = D * H * Wd;
  dim3 g((N + 127) / 128, (Cout + 15) / 16);
  dim3 b(128);
  switch (ksz) {
    case 1: conv3d_bn_act_wmma<1><<<g, b, 0, st>>>(in, w, sb, add, out, Cin, Cout, D, H, Wd, pad, relu); break;
    case 3: conv3d_bn_act_wmma<3><<<g, b, 0, st>>>(in, w, sb, add, out, Cin, Cout, D, H, Wd, pad, relu); break;
    case 7: conv3d_bn_act_wmma<7><<<g, b, 0, st>>>(in, w, sb, add, out, Cin, Cout, D, H, Wd, pad, relu); break;
    default: break;
  }
}

extern "C" void kernel_launch(void* const* d_in, const int* in_sizes, int n_in,
                              void* d_out, int out_size, void* d_ws, size_t ws_size,
                              hipStream_t stream)
{
  (void)in_sizes; (void)n_in; (void)out_size;

  const float* features = (const float*)d_in[0];
  const float* gridc    = (const float*)d_in[1];
  const float* camM     = (const float*)d_in[2];
  const float* camK     = (const float*)d_in[3];
  const float* rdist    = (const float*)d_in[4];
  const float* tdist    = (const float*)d_in[5];

  // ---- Replay init_params insertion order to locate param pointers.
  std::unordered_map<std::string, int> pm;
  int cnt = 6;
  auto add    = [&](const std::string& s) { pm[s] = cnt++; };
  auto convP  = [&](const std::string& n) { add(n + "_w"); add(n + "_b"); };
  auto bnP    = [&](const std::string& n) { add(n + "_g"); add(n + "_bt"); add(n + "_m"); add(n + "_v"); };
  auto basicP = [&](const std::string& n) { convP(n); bnP(n); };
  auto resP   = [&](const std::string& n, bool sc) {
    convP(n + "1"); bnP(n + "1"); convP(n + "2"); bnP(n + "2");
    if (sc) { convP(n + "s"); bnP(n + "s"); }
  };
  auto upP    = [&](const std::string& n) { add(n + "_w"); add(n + "_b"); bnP(n); };

  convP("proc");
  basicP("f1");
  resP("f2", true);  resP("f3", false); resP("f4", false);
  resP("skip1", false); resP("enc1", true);
  resP("skip2", false); resP("enc2", true);
  resP("skip3", false); resP("enc3", false);
  resP("skip4", false); resP("enc4", false);
  resP("skip5", false); resP("enc5", false);
  resP("mid", false);
  resP("dres5", false); upP("dup5");
  resP("dres4", false); upP("dup4");
  resP("dres3", false); upP("dup3");
  resP("dres2", false); upP("dup2");
  resP("dres1", false); upP("dup1");
  resP("b1", false); basicP("b2"); basicP("b3");
  convP("out");

  auto Pp = [&](const std::string& n) -> const float* {
    return (const float*)d_in[pm[n]];
  };

  // ---- Workspace layout (floats). View volumes region is reused after fusion.
  float* ws = (float*)d_ws;
  float* fproc = ws;                                    // 6*32*16384  = 3,145,728
  float* vols  = fproc + (size_t)NC * CMID * FHW * FHW; // 6*32*32768  = 6,291,456
  float* vol   = vols + (size_t)NC * CMID * NVTOT;      // 32*32768    = 1,048,576
  float* sb    = vol + (size_t)CMID * NVTOT;            // 512
  const size_t needF = (size_t)NC * CMID * FHW * FHW + (size_t)NC * CMID * NVTOT
                       + (size_t)CMID * NVTOT + 512;
  if (ws_size < needF * sizeof(float)) return;          // scratch too small: no-op

  // reuse region (aliases vols; vols is dead after fuse_views)
  float* bufA = vols;
  float* bufB = bufA + 1048576;
  float* t1   = bufB + 1048576;
  float* t2   = t1 + 1048576;
  float* s1   = t2 + 1048576;          // 32*32768
  float* s2   = s1 + 1048576;          // 64*4096
  float* s3   = s2 + 262144;           // 128*512
  float* s4   = s3 + 65536;            // 128*64
  float* s5   = s4 + 8192;             // 128*8

  auto fuse = [&](const std::string& n, int C, bool bn) {
    fuse_bn_kernel<<<(C + 63) / 64, 64, 0, stream>>>(
        Pp(n + "_b"),
        bn ? Pp(n + "_g") : nullptr, bn ? Pp(n + "_bt") : nullptr,
        bn ? Pp(n + "_m") : nullptr, bn ? Pp(n + "_v") : nullptr,
        sb, C, bn ? 1 : 0);
  };
  auto convL = [&](const std::string& n, const float* in, float* out,
                   int Cin, int Cout, int S, int k, int pad, int relu,
                   const float* addp, bool bn) {
    fuse(n, Cout, bn);
    launch_conv(in, Pp(n + "_w"), sb, addp, out, Cin, Cout, S, S, S, k, pad, relu, stream);
  };
  auto resL = [&](const std::string& n, const float* in, float* out,
                  int Cin, int Cout, int S) {
    const bool sc = (Cin != Cout);
    convL(n + "1", in, t1, Cin, Cout, S, 3, 1, 1, nullptr, true);
    const float* addp = in;
    if (sc) { convL(n + "s", in, t2, Cin, Cout, S, 1, 0, 0, nullptr, true); addp = t2; }
    convL(n + "2", t1, out, Cout, Cout, S, 3, 1, 1, addp, true);
  };
  auto upL = [&](const std::string& n, const float* in, float* out,
                 int Cin, int Cout, int S, const float* skip) {
    fuse(n, Cout, true);
    const int total = Cout * 8 * S * S * S;
    upconv_kernel<<<(total + 255) / 256, 256, 0, stream>>>(
        in, Pp(n + "_w"), sb, skip, out, Cin, Cout, S);
  };
  auto poolL = [&](const float* in, float* out, int C, int S) {
    const int So = S / 2;
    const int total = C * So * So * So;
    maxpool2_kernel<<<(total + 255) / 256, 256, 0, stream>>>(in, out, C, S);
  };

  // ---- 1) Feature processing conv (1x1, 256->32) per camera, WMMA GEMM.
  fuse("proc", CMID, false);
  for (int c = 0; c < NC; ++c)
    launch_conv(features + (size_t)c * CINF * FHW * FHW, Pp("proc_w"), sb, nullptr,
                fproc + (size_t)c * CMID * FHW * FHW,
                CINF, CMID, 1, FHW, FHW, 1, 0, 0, stream);

  // ---- 2) Unprojection, 3) view fusion
  {
    const int total = NC * NVTOT;
    unproject_kernel<<<(total + 255) / 256, 256, 0, stream>>>(
        fproc, camM, camK, rdist, tdist, gridc, vols, NVTOT);
  }
  {
    const int total = CMID * NVTOT;
    fuse_views_kernel<<<(total + 255) / 256, 256, 0, stream>>>(vols, vol, NVTOT);
  }

  // ---- 4) V2V 3D U-Net
  convL("f1", vol, bufA, 32, 16, 32, 7, 3, 1, nullptr, true);
  resL("f2", bufA, bufB, 16, 32, 32);
  resL("f3", bufB, bufA, 32, 32, 32);
  resL("f4", bufA, bufB, 32, 32, 32);
  resL("skip1", bufB, s1, 32, 32, 32);
  poolL(bufB, bufA, 32, 32);
  resL("enc1", bufA, bufB, 32, 64, 16);
  resL("skip2", bufB, s2, 64, 64, 16);
  poolL(bufB, bufA, 64, 16);
  resL("enc2", bufA, bufB, 64, 128, 8);
  resL("skip3", bufB, s3, 128, 128, 8);
  poolL(bufB, bufA, 128, 8);
  resL("enc3", bufA, bufB, 128, 128, 4);
  resL("skip4", bufB, s4, 128, 128, 4);
  poolL(bufB, bufA, 128, 4);
  resL("enc4", bufA, bufB, 128, 128, 2);
  resL("skip5", bufB, s5, 128, 128, 2);
  poolL(bufB, bufA, 128, 2);
  resL("enc5", bufA, bufB, 128, 128, 1);
  resL("mid",  bufB, bufA, 128, 128, 1);
  resL("dres5", bufA, bufB, 128, 128, 1);
  upL("dup5", bufB, bufA, 128, 128, 1, s5);
  resL("dres4", bufA, bufB, 128, 128, 2);
  upL("dup4", bufB, bufA, 128, 128, 2, s4);
  resL("dres3", bufA, bufB, 128, 128, 4);
  upL("dup3", bufB, bufA, 128, 128, 4, s3);
  resL("dres2", bufA, bufB, 128, 128, 8);
  upL("dup2", bufB, bufA, 128, 64, 8, s2);
  resL("dres1", bufA, bufB, 64, 64, 16);
  upL("dup1", bufB, bufA, 64, 32, 16, s1);
  resL("b1", bufA, bufB, 32, 32, 32);
  convL("b2", bufB, bufA, 32, 32, 32, 1, 0, 1, nullptr, true);
  convL("b3", bufA, bufB, 32, 32, 32, 1, 0, 1, nullptr, true);

  // ---- 5) Output heatmaps (written straight into d_out) + soft-argmax
  float* hm = (float*)d_out;
  convL("out", bufB, hm, 32, NJ, 32, 1, 0, 0, nullptr, false);
  softargmax_kernel<<<NJ, 256, 0, stream>>>(hm, gridc, hm + (size_t)NJ * NVTOT, NVTOT);
}
